// MultiHeadAttention_84490596647332
// MI455X (gfx1250) — matrix operand order repaired and checked by reading, not compiled, not run on hardware
//
#include <hip/hip_runtime.h>

// MI455X / gfx1250 multi-head attention forward (fp32 ref, bf16 WMMA compute).
// B=2, N=2048, C=2048, H=16, hd=128, RoPE base 10000, non-causal softmax.

typedef __bf16 bf16;
typedef __attribute__((ext_vector_type(8)))  __bf16 v8bf;
typedef __attribute__((ext_vector_type(16))) __bf16 v16bf;
typedef __attribute__((ext_vector_type(8)))  float  v8f;

#define BATCH   2
#define SEQ     2048
#define CDIM    2048
#define HEADS   16
#define HD      128
#define QKVDIM  (3 * CDIM)          // 6144
#define ROWS    (BATCH * SEQ)       // 4096

// ---------------------------------------------------------------------------
// WMMA helpers (layouts per CDNA5 ISA §7.12.2)
// ---------------------------------------------------------------------------
__device__ __forceinline__ v8f wmma_bf16f32(v16bf a, v16bf b, v8f c) {
  // v_wmma_f32_16x16x32_bf16  (probe-confirmed 8-arg signature)
  return __builtin_amdgcn_wmma_f32_16x16x32_bf16(
      /*neg_a=*/false, a, /*neg_b=*/false, b,
      /*c_mod=*/(short)0, c, /*reuse_a=*/false, /*reuse_b=*/false);
}

// A fragment: 16x32 bf16, rows m0..m0+15, K-chunk [k0, k0+32).
// Lane (m = m0 + lane&15, g = lane>>4) holds K = {8g..8g+7} ∪ {16+8g..16+8g+7}.
__device__ __forceinline__ v16bf load_fragA(const bf16* A, int lda, int m0,
                                            int k0, int lane) {
  const int m = m0 + (lane & 15);
  const int g = lane >> 4;
  const bf16* p = A + (size_t)m * lda + k0;
  v8bf lo = *(const v8bf*)(p + 8 * g);        // K = 8g .. 8g+7
  v8bf hi = *(const v8bf*)(p + 16 + 8 * g);   // K = 16+8g .. 16+8g+7
  return __builtin_shufflevector(lo, hi, 0, 1, 2, 3, 4, 5, 6, 7,
                                 8, 9, 10, 11, 12, 13, 14, 15);
}

// B fragment: 32x16 bf16 (K x N), columns n0..n0+15, K-chunk [k0, k0+32).
// Bt is stored column-major-as-rows: Bt[n][k] with K contiguous (ldb elems).
// Lane (n = n0 + lane&15, g = lane>>4) holds K = {8g..8g+7} ∪ {16+8g..16+8g+7}.
__device__ __forceinline__ v16bf load_fragB(const bf16* Bt, int ldb, int n0,
                                            int k0, int lane) {
  const int n = n0 + (lane & 15);
  const int g = lane >> 4;
  const bf16* p = Bt + (size_t)n * ldb + k0;
  v8bf lo = *(const v8bf*)(p + 8 * g);
  v8bf hi = *(const v8bf*)(p + 16 + 8 * g);
  return __builtin_shufflevector(lo, hi, 0, 1, 2, 3, 4, 5, 6, 7,
                                 8, 9, 10, 11, 12, 13, 14, 15);
}

// Row reductions across the 16 lanes of a half-wave (C/D row lives in one
// register index across lanes 0-15 / 16-31).
__device__ __forceinline__ float redmax16(float v) {
  v = fmaxf(v, __shfl_xor(v, 1, 32));
  v = fmaxf(v, __shfl_xor(v, 2, 32));
  v = fmaxf(v, __shfl_xor(v, 4, 32));
  v = fmaxf(v, __shfl_xor(v, 8, 32));
  return v;
}
__device__ __forceinline__ float redsum16(float v) {
  v += __shfl_xor(v, 1, 32);
  v += __shfl_xor(v, 2, 32);
  v += __shfl_xor(v, 4, 32);
  v += __shfl_xor(v, 8, 32);
  return v;
}

// ---------------------------------------------------------------------------
// Precision-staging kernels
// ---------------------------------------------------------------------------
__global__ void cvt_f32_bf16_kernel(const float* __restrict__ src,
                                    bf16* __restrict__ dst, int total) {
  int i = blockIdx.x * blockDim.x + threadIdx.x;
  if (i < total) dst[i] = (bf16)src[i];
}

// src: [K][N] fp32 row-major  ->  dst: [N][K] bf16 (K contiguous per row)
__global__ void transpose_bf16_kernel(const float* __restrict__ src,
                                      bf16* __restrict__ dst, int K, int N) {
  size_t i = (size_t)blockIdx.x * blockDim.x + threadIdx.x;
  size_t total = (size_t)K * N;
  if (i >= total) return;
  int n = (int)(i / K);
  int k = (int)(i % K);
  dst[i] = (bf16)src[(size_t)k * N + n];
}

// ---------------------------------------------------------------------------
// GEMM 1: qkv = x @ W_qkv ; scatter into Q/K [B,H,N,hd] bf16 and Vt [B,H,hd,N]
// Wave tile: 32(M) x 64(N); block = 4 waves stacked in M (128 rows x 64 cols).
// ---------------------------------------------------------------------------
__global__ __launch_bounds__(128) void qkv_gemm_kernel(
    const bf16* __restrict__ Xb, const bf16* __restrict__ WqkvT,
    bf16* __restrict__ Qraw, bf16* __restrict__ Kraw, bf16* __restrict__ Vt) {
  const int lane = threadIdx.x & 31;
  const int wave = threadIdx.x >> 5;
  const int m0 = blockIdx.x * 128 + wave * 32;
  const int n0 = blockIdx.y * 64;

  v8f acc[2][4];
#pragma unroll
  for (int mi = 0; mi < 2; ++mi)
#pragma unroll
    for (int j = 0; j < 4; ++j) acc[mi][j] = (v8f){0.f, 0.f, 0.f, 0.f, 0.f, 0.f, 0.f, 0.f};

  for (int k0 = 0; k0 < CDIM; k0 += 32) {
    if (k0 + 32 < CDIM) {  // uniform branch; CDNA5 global_prefetch_b8
      __builtin_prefetch(Xb + (size_t)(m0 + (lane & 15)) * CDIM + k0 + 32, 0, 3);
      __builtin_prefetch(WqkvT + (size_t)(n0 + (lane & 15)) * CDIM + k0 + 32, 0, 3);
    }
    v16bf a0 = load_fragA(Xb, CDIM, m0, k0, lane);
    v16bf a1 = load_fragA(Xb, CDIM, m0 + 16, k0, lane);
#pragma unroll
    for (int j = 0; j < 4; ++j) {
      v16bf bj = load_fragB(WqkvT, CDIM, n0 + 16 * j, k0, lane);
      acc[0][j] = wmma_bf16f32(a0, bj, acc[0][j]);
      acc[1][j] = wmma_bf16f32(a1, bj, acc[1][j]);
    }
  }

  const int g = lane >> 4, c = lane & 15;
#pragma unroll
  for (int mi = 0; mi < 2; ++mi)
#pragma unroll
    for (int j = 0; j < 4; ++j)
#pragma unroll
      for (int r = 0; r < 8; ++r) {
        int grow = m0 + mi * 16 + r + 8 * g;   // b*2048 + n
        int gcol = n0 + 16 * j + c;            // within [0, 6144)
        float val = acc[mi][j][r];
        int which = gcol >> 11;                // 0=q 1=k 2=v
        int rem = gcol & 2047;
        int h = rem >> 7, d = rem & 127;
        int b = grow >> 11, n = grow & 2047;
        size_t bh = (size_t)(b * HEADS + h);
        if (which == 0)
          Qraw[(bh * SEQ + n) * HD + d] = (bf16)val;
        else if (which == 1)
          Kraw[(bh * SEQ + n) * HD + d] = (bf16)val;
        else
          Vt[(bh * HD + d) * SEQ + n] = (bf16)val;   // transposed for PV B-matrix
      }
}

// ---------------------------------------------------------------------------
// RoPE (half-split), fp32 math, bf16 in/out, in-place. One thread per (bh,n,i)
// pair handles elements i and i+64. blockIdx.y: 0 -> Q, 1 -> K.
// ---------------------------------------------------------------------------
__global__ void rope_kernel(bf16* __restrict__ Qraw, bf16* __restrict__ Kraw) {
  int idx = blockIdx.x * blockDim.x + threadIdx.x;  // [0, 32*2048*64)
  bf16* buf = blockIdx.y ? Kraw : Qraw;
  int i = idx & 63;
  int n = (idx >> 6) & 2047;
  int bh = idx >> 17;
  // inv_freq = 10000^(-i/64) = exp(-ln(10000) * i / 64)
  float inv_f = __expf(-9.210340371976184f * (float)i * (1.0f / 64.0f));
  float ang = (float)n * inv_f;
  float s, c;
  __sincosf(ang, &s, &c);
  size_t base = ((size_t)bh * SEQ + n) * HD;
  float x1 = (float)buf[base + i];
  float x2 = (float)buf[base + 64 + i];
  buf[base + i]      = (bf16)(x1 * c - x2 * s);
  buf[base + 64 + i] = (bf16)(x1 * s + x2 * c);
}

// ---------------------------------------------------------------------------
// Flash attention: block = 4 waves; wave owns 16 queries, streams 32 keys/iter.
// Online softmax in fp32; P re-laid out C-frag -> A-frag via per-wave LDS.
// Output written as bf16 A-matrix rows [b*2048+n][h*128+d] for the out-proj.
// ---------------------------------------------------------------------------
__global__ __launch_bounds__(128) void attn_kernel(
    const bf16* __restrict__ Qb, const bf16* __restrict__ Kb,
    const bf16* __restrict__ Vt, bf16* __restrict__ attnA) {
  __shared__ __align__(128) bf16 sP[4 * 16 * 32];  // 1KB per wave
  const int lane = threadIdx.x & 31;
  const int wave = threadIdx.x >> 5;
  const int bh = blockIdx.x >> 5;  // 0..31  (b*16+h)
  const int qt = blockIdx.x & 31;  // 0..31  (64-query tiles)
  const int q0 = qt * 64 + wave * 16;
  const bf16* Q = Qb + (size_t)bh * SEQ * HD;
  const bf16* K = Kb + (size_t)bh * SEQ * HD;
  const bf16* V = Vt + (size_t)bh * HD * SEQ;
  bf16* myP = sP + wave * 512;
  const float scale = 0.08838834764831845f;  // 1/sqrt(128)
  const int g = lane >> 4, cidx = lane & 15;

  v16bf aq[4];
#pragma unroll
  for (int ck = 0; ck < 4; ++ck) aq[ck] = load_fragA(Q, HD, q0, 32 * ck, lane);

  v8f acc[8];
  float mrow[8], lrow[8];
#pragma unroll
  for (int t = 0; t < 8; ++t) acc[t] = (v8f){0.f, 0.f, 0.f, 0.f, 0.f, 0.f, 0.f, 0.f};
#pragma unroll
  for (int r = 0; r < 8; ++r) { mrow[r] = -1e30f; lrow[r] = 0.f; }

  for (int k0 = 0; k0 < SEQ; k0 += 32) {
    // S = Q * K^T for 32 keys (two 16-col tiles), contraction over hd=128
    v8f s0 = (v8f){0.f, 0.f, 0.f, 0.f, 0.f, 0.f, 0.f, 0.f};
    v8f s1 = (v8f){0.f, 0.f, 0.f, 0.f, 0.f, 0.f, 0.f, 0.f};
#pragma unroll
    for (int ck = 0; ck < 4; ++ck) {
      v16bf b0 = load_fragB(K, HD, k0, 32 * ck, lane);
      v16bf b1 = load_fragB(K, HD, k0 + 16, 32 * ck, lane);
      s0 = wmma_bf16f32(aq[ck], b0, s0);
      s1 = wmma_bf16f32(aq[ck], b1, s1);
    }
    // Online softmax per row (row r+8g lives in register r across 16 lanes)
#pragma unroll
    for (int r = 0; r < 8; ++r) {
      float v0 = s0[r] * scale, v1 = s1[r] * scale;
      float rm = redmax16(fmaxf(v0, v1));
      float nm = fmaxf(mrow[r], rm);
      float corr = __expf(mrow[r] - nm);
      float p0 = __expf(v0 - nm);
      float p1 = __expf(v1 - nm);
      lrow[r] = lrow[r] * corr + redsum16(p0 + p1);
      mrow[r] = nm;
      int row = r + 8 * g;
      myP[row * 32 + cidx]      = (bf16)p0;
      myP[row * 32 + 16 + cidx] = (bf16)p1;
#pragma unroll
      for (int t = 0; t < 8; ++t) acc[t][r] *= corr;
    }
    // Per-wave LDS region; DS ops are in-order per wave — wait for stores,
    // then reload P in A-fragment layout.
    asm volatile("s_wait_dscnt 0x0" ::: "memory");
    v16bf ap = load_fragA(myP, 32, 0, 0, lane);
#pragma unroll
    for (int t = 0; t < 8; ++t) {
      v16bf bv = load_fragB(V, SEQ, t * 16, k0, lane);  // V^T: dims x keys
      acc[t] = wmma_bf16f32(ap, bv, acc[t]);
    }
  }

  // Normalize and emit bf16 rows for the output projection
  const int b = bh >> 4, h = bh & 15;
#pragma unroll
  for (int r = 0; r < 8; ++r) {
    float inv = 1.0f / lrow[r];
    int n = q0 + r + 8 * g;
    size_t rowbase = ((size_t)b * SEQ + n) * CDIM + (size_t)h * HD;
#pragma unroll
    for (int t = 0; t < 8; ++t)
      attnA[rowbase + t * 16 + cidx] = (bf16)(acc[t][r] * inv);
  }
}

// ---------------------------------------------------------------------------
// GEMM 2: out = attn @ W_out  (fp32 output straight into d_out [B,N,C])
// ---------------------------------------------------------------------------
__global__ __launch_bounds__(128) void out_gemm_kernel(
    const bf16* __restrict__ Ab, const bf16* __restrict__ WoutT,
    float* __restrict__ out) {
  const int lane = threadIdx.x & 31;
  const int wave = threadIdx.x >> 5;
  const int m0 = blockIdx.x * 128 + wave * 32;
  const int n0 = blockIdx.y * 64;

  v8f acc[2][4];
#pragma unroll
  for (int mi = 0; mi < 2; ++mi)
#pragma unroll
    for (int j = 0; j < 4; ++j) acc[mi][j] = (v8f){0.f, 0.f, 0.f, 0.f, 0.f, 0.f, 0.f, 0.f};

  for (int k0 = 0; k0 < CDIM; k0 += 32) {
    if (k0 + 32 < CDIM) {
      __builtin_prefetch(Ab + (size_t)(m0 + (lane & 15)) * CDIM + k0 + 32, 0, 3);
      __builtin_prefetch(WoutT + (size_t)(n0 + (lane & 15)) * CDIM + k0 + 32, 0, 3);
    }
    v16bf a0 = load_fragA(Ab, CDIM, m0, k0, lane);
    v16bf a1 = load_fragA(Ab, CDIM, m0 + 16, k0, lane);
#pragma unroll
    for (int j = 0; j < 4; ++j) {
      v16bf bj = load_fragB(WoutT, CDIM, n0 + 16 * j, k0, lane);
      acc[0][j] = wmma_bf16f32(a0, bj, acc[0][j]);
      acc[1][j] = wmma_bf16f32(a1, bj, acc[1][j]);
    }
  }

  const int g = lane >> 4, c = lane & 15;
#pragma unroll
  for (int mi = 0; mi < 2; ++mi)
#pragma unroll
    for (int j = 0; j < 4; ++j)
#pragma unroll
      for (int r = 0; r < 8; ++r) {
        int grow = m0 + mi * 16 + r + 8 * g;
        int gcol = n0 + 16 * j + c;
        out[(size_t)grow * CDIM + gcol] = acc[mi][j][r];
      }
}

// ---------------------------------------------------------------------------
// Launcher
// ---------------------------------------------------------------------------
extern "C" void kernel_launch(void* const* d_in, const int* in_sizes, int n_in,
                              void* d_out, int out_size, void* d_ws,
                              size_t ws_size, hipStream_t stream) {
  const float* x     = (const float*)d_in[0];  // [B,N,C]
  const float* W_qkv = (const float*)d_in[1];  // [C, 3C]
  const float* W_out = (const float*)d_in[2];  // [C, C]
  float* out = (float*)d_out;                  // [B,N,C]

  // Workspace carve-up (all sizes are multiples of 256B); total ~112 MB.
  char* p = (char*)d_ws;
  const size_t SZ_XB  = (size_t)ROWS * CDIM * sizeof(bf16);            // 16 MB
  const size_t SZ_WQT = (size_t)QKVDIM * CDIM * sizeof(bf16);          // 24 MB
  const size_t SZ_WOT = (size_t)CDIM * CDIM * sizeof(bf16);            //  8 MB
  const size_t SZ_HED = (size_t)BATCH * HEADS * SEQ * HD * sizeof(bf16);// 16 MB
  bf16* xb    = (bf16*)p; p += SZ_XB;
  bf16* wqT   = (bf16*)p; p += SZ_WQT;
  bf16* woT   = (bf16*)p; p += SZ_WOT;
  bf16* Qraw  = (bf16*)p; p += SZ_HED;
  bf16* Kraw  = (bf16*)p; p += SZ_HED;
  bf16* Vt    = (bf16*)p; p += SZ_HED;
  bf16* attnA = (bf16*)p; p += SZ_XB;

  // 1. Precision staging
  {
    int total = ROWS * CDIM;  // 8,388,608
    cvt_f32_bf16_kernel<<<(total + 255) / 256, 256, 0, stream>>>(x, xb, total);
  }
  {
    size_t total = (size_t)CDIM * QKVDIM;  // 12,582,912
    transpose_bf16_kernel<<<(unsigned)((total + 255) / 256), 256, 0, stream>>>(
        W_qkv, wqT, CDIM, QKVDIM);
  }
  {
    size_t total = (size_t)CDIM * CDIM;  // 4,194,304
    transpose_bf16_kernel<<<(unsigned)((total + 255) / 256), 256, 0, stream>>>(
        W_out, woT, CDIM, CDIM);
  }

  // 2. QKV projection (M=4096, N=6144, K=2048)
  qkv_gemm_kernel<<<dim3(ROWS / 128, QKVDIM / 64), 128, 0, stream>>>(
      xb, wqT, Qraw, Kraw, Vt);

  // 3. RoPE on Q and K
  {
    int pairs = BATCH * HEADS * SEQ * (HD / 2);  // 4,194,304
    rope_kernel<<<dim3(pairs / 256, 2), 256, 0, stream>>>(Qraw, Kraw);
  }

  // 4. Flash attention: 2*16 (b,h) x 32 query tiles of 64
  attn_kernel<<<BATCH * HEADS * (SEQ / 64), 128, 0, stream>>>(Qraw, Kraw, Vt,
                                                              attnA);

  // 5. Output projection (M=4096, N=2048, K=2048) -> fp32 d_out
  out_gemm_kernel<<<dim3(ROWS / 128, CDIM / 64), 128, 0, stream>>>(attnA, woT,
                                                                   out);
}
